// InputMapCTRNN_45896020525302
// MI455X (gfx1250) — matrix operand order repaired and checked by reading, not compiled, not run on hardware
//
#include <hip/hip_runtime.h>

// CDNA5 / gfx1250 — wave32, bf16 WMMA (16x16x32, f32 accum).
// Split-K + software-pipelined fragment loads with compile-time trip counts.

typedef __attribute__((ext_vector_type(16))) __bf16 v16bf;
typedef __attribute__((ext_vector_type(8)))  float  v8f;

constexpr int T_ = 256, B_ = 128, I_ = 64, C_ = 256, H_ = 512, O_ = 128;
constexpr int KH_ = C_ * I_ + I_;       // 16448 real K: kron(ctx,x) + x (bias rows)
constexpr int KP_ = 16896;              // padded K = 16 * 1056 (zero tail)
constexpr int KSPLIT_ = 16;             // split-K ways
constexpr int SPLITLEN_ = KP_ / KSPLIT_;  // 1056 elements = 33 k-steps, constant
constexpr int KC_ = C_ + I_;            // 320: [ctx ; x] features for ctx update

#define DEV static __device__ __forceinline__

DEV unsigned short f2bf(float f) {  // round-to-nearest-even fp32 -> bf16
  unsigned int u = __float_as_uint(f);
  u += 0x7FFFu + ((u >> 16) & 1u);
  return (unsigned short)(u >> 16);
}
DEV float bf2f(unsigned short h) { return __uint_as_float(((unsigned int)h) << 16); }

union Frag { v16bf v; uint4 q[2]; };

DEV void zero_acc(v8f acc[4]) {
#pragma unroll
  for (int j = 0; j < 4; ++j)
#pragma unroll
    for (int e = 0; e < 8; ++e) acc[j][e] = 0.0f;
}

// One wave computes a 16(M) x 64(N) f32 tile over KLEN (compile-time, mult 32).
// A: row-major bf16 [M,K]; B: N-major bf16 [N,K] (each output column's
// K-vector contiguous). Fragment->lane mapping (contiguous k order, same for A and B):
//  A: lane=(m|hl<<4): elems e    -> K=k0+16*hl+e, row m.
//  B: lane=(n|hl<<4): elems e    -> K=k0+16*hl+e, column n.
// Pipelined (prefetch distance 1): each WMMA waits only on loads issued one
// iteration earlier; constant trip count lets unrolling rename the buffers.
template <int KLEN>
DEV void wave_tile_16x64(const unsigned short* __restrict__ Abase, int lda,
                         const unsigned short* __restrict__ Bbase, int ldb,
                         v8f acc[4]) {
  const int lane = threadIdx.x & 31;
  const int hl = lane >> 4;
  const int ml = lane & 15;
  const unsigned short* arow = Abase + (size_t)ml * lda + 16 * hl;
  const unsigned short* br0 = Bbase + (size_t)ml * ldb + 16 * hl;
  const unsigned short* br1 = br0 + (size_t)16 * ldb;
  const unsigned short* br2 = br0 + (size_t)32 * ldb;
  const unsigned short* br3 = br0 + (size_t)48 * ldb;

  Frag a, b0, b1, b2, b3;
  a.q[0] = *(const uint4*)(arow);
  a.q[1] = *(const uint4*)(arow + 8);
  b0.q[0] = *(const uint4*)(br0); b0.q[1] = *(const uint4*)(br0 + 8);
  b1.q[0] = *(const uint4*)(br1); b1.q[1] = *(const uint4*)(br1 + 8);
  b2.q[0] = *(const uint4*)(br2); b2.q[1] = *(const uint4*)(br2 + 8);
  b3.q[0] = *(const uint4*)(br3); b3.q[1] = *(const uint4*)(br3 + 8);

#pragma unroll 2
  for (int k = 32; k < KLEN; k += 32) {
    Frag an, bn0, bn1, bn2, bn3;
    an.q[0] = *(const uint4*)(arow + k);
    an.q[1] = *(const uint4*)(arow + k + 8);
    bn0.q[0] = *(const uint4*)(br0 + k); bn0.q[1] = *(const uint4*)(br0 + k + 8);
    bn1.q[0] = *(const uint4*)(br1 + k); bn1.q[1] = *(const uint4*)(br1 + k + 8);
    bn2.q[0] = *(const uint4*)(br2 + k); bn2.q[1] = *(const uint4*)(br2 + k + 8);
    bn3.q[0] = *(const uint4*)(br3 + k); bn3.q[1] = *(const uint4*)(br3 + k + 8);
    acc[0] = __builtin_amdgcn_wmma_f32_16x16x32_bf16(false, a.v, false, b0.v, (short)0, acc[0], false, false);
    acc[1] = __builtin_amdgcn_wmma_f32_16x16x32_bf16(false, a.v, false, b1.v, (short)0, acc[1], false, false);
    acc[2] = __builtin_amdgcn_wmma_f32_16x16x32_bf16(false, a.v, false, b2.v, (short)0, acc[2], false, false);
    acc[3] = __builtin_amdgcn_wmma_f32_16x16x32_bf16(false, a.v, false, b3.v, (short)0, acc[3], false, false);
    a = an; b0 = bn0; b1 = bn1; b2 = bn2; b3 = bn3;
  }
  acc[0] = __builtin_amdgcn_wmma_f32_16x16x32_bf16(false, a.v, false, b0.v, (short)0, acc[0], false, false);
  acc[1] = __builtin_amdgcn_wmma_f32_16x16x32_bf16(false, a.v, false, b1.v, (short)0, acc[1], false, false);
  acc[2] = __builtin_amdgcn_wmma_f32_16x16x32_bf16(false, a.v, false, b2.v, (short)0, acc[2], false, false);
  acc[3] = __builtin_amdgcn_wmma_f32_16x16x32_bf16(false, a.v, false, b3.v, (short)0, acc[3], false, false);
}

// 16x16 tile (single accumulator) for the fused output GEMM; A lives in LDS
// (ds_load_b128 fragments after inlining).
template <int KLEN>
DEV void wave_tile_16x16(const unsigned short* Abase, int lda,
                         const unsigned short* __restrict__ Bbase, int ldb,
                         v8f& acc) {
  const int lane = threadIdx.x & 31;
  const int hl = lane >> 4;
  const int ml = lane & 15;
  const unsigned short* arow = Abase + (size_t)ml * lda + 16 * hl;
  const unsigned short* brow = Bbase + (size_t)ml * ldb + 16 * hl;
#pragma unroll 4
  for (int k = 0; k < KLEN; k += 32) {
    Frag a, b;
    a.q[0] = *(const uint4*)(arow + k);
    a.q[1] = *(const uint4*)(arow + k + 8);
    b.q[0] = *(const uint4*)(brow + k);
    b.q[1] = *(const uint4*)(brow + k + 8);
    acc = __builtin_amdgcn_wmma_f32_16x16x32_bf16(false, a.v, false, b.v, (short)0, acc, false, false);
  }
}

// ---- One-time weight pack: fp32 -> bf16, Wmap permuted to N-major kron form,
//      K zero-padded to KP_.
__global__ __launch_bounds__(256) void pack_kernel(
    const float* __restrict__ Wcc, const float* __restrict__ Wic,
    const float* __restrict__ Wmap, const float* __restrict__ bmap,
    const float* __restrict__ Whh, const float* __restrict__ Who,
    unsigned short* __restrict__ Wth, unsigned short* __restrict__ Wctx,
    unsigned short* __restrict__ WhhP, unsigned short* __restrict__ WhoP,
    float* __restrict__ ctx) {
  int idx = blockIdx.x * 256 + (int)threadIdx.x;
  const int nWth = H_ * KP_, nWctx = C_ * KC_, nWhh = H_ * H_, nWho = O_ * H_;
  if (idx < nWth) {
    int hh = idx / KP_, k = idx - hh * KP_;
    float v = 0.0f;
    if (k < C_ * I_)
      v = Wmap[(size_t)(hh * I_ + (k & (I_ - 1))) * C_ + (k >> 6)];
    else if (k < KH_)
      v = bmap[hh * I_ + (k - C_ * I_)];
    Wth[idx] = f2bf(v);
    return;
  }
  idx -= nWth;
  if (idx < nWctx) {
    int co = idx / KC_, k = idx - co * KC_;
    float v = (k < C_) ? Wcc[co * C_ + k] : Wic[co * I_ + (k - C_)];
    Wctx[idx] = f2bf(v);
    return;
  }
  idx -= nWctx;
  if (idx < nWhh) { WhhP[idx] = f2bf(Whh[idx]); return; }
  idx -= nWhh;
  if (idx < nWho) { WhoP[idx] = f2bf(Who[idx]); return; }
  idx -= nWho;
  if (idx < B_ * C_) ctx[idx] = 0.0f;  // ctx0 = zeros
}

// ---- Per-step feature build: f[b, c*I+i] = ctx*x kron; then x (bias rows);
//      zero pad to KP_. fctx[b,:] = [ctx ; x] bf16 for the ctx-update GEMM.
__global__ __launch_bounds__(256) void feat_kernel(
    const float* __restrict__ ctx, const float* __restrict__ xt,
    unsigned short* __restrict__ f, unsigned short* __restrict__ fctx) {
  int idx = blockIdx.x * 256 + (int)threadIdx.x;
  const int nF = B_ * KP_;
  if (idx < nF) {
    int b = idx / KP_, k = idx - b * KP_;
    float v = 0.0f;
    if (k < C_ * I_)
      v = ctx[b * C_ + (k >> 6)] * xt[b * I_ + (k & 63)];
    else if (k < KH_)
      v = xt[b * I_ + (k - C_ * I_)];
    f[idx] = f2bf(v);
    return;
  }
  idx -= nF;
  if (idx < B_ * KC_) {
    int b = idx / KC_, k = idx - b * KC_;
    float v = (k < C_) ? ctx[b * C_ + k] : xt[b * I_ + (k - C_)];
    fctx[idx] = f2bf(v);
  }
}

// ---- Split-K hypernet GEMM: hpart[s] = partial f @ Wth^T over K-chunk s.
// M=128, N=512, K=16896(padded), 16 splits x 64 tiles = 1024 waves, 128 blocks.
__global__ __launch_bounds__(256) void gemm_h_kernel(
    const unsigned short* __restrict__ f, const unsigned short* __restrict__ Wth,
    float* __restrict__ hpart) {
  int wave = __builtin_amdgcn_readfirstlane((blockIdx.x * 256 + (int)threadIdx.x) >> 5);
  int s = wave >> 6;           // split 0..15
  int t = wave & 63;           // tile within split
  int nb = t & 7, mb = t >> 3; // 8x8 tiles of 16x64
  int kBeg = s * SPLITLEN_;
  v8f acc[4];
  zero_acc(acc);
  wave_tile_16x64<SPLITLEN_>(f + (size_t)(mb * 16) * KP_ + kBeg, KP_,
                             Wth + (size_t)(nb * 64) * KP_ + kBeg, KP_, acc);
  int lane = threadIdx.x & 31, hl = lane >> 4, ml = lane & 15;
  float* dst = hpart + (size_t)s * B_ * H_;
#pragma unroll
  for (int j = 0; j < 4; ++j)
#pragma unroll
    for (int r = 0; r < 8; ++r) {
      int m = mb * 16 + r + 8 * hl;
      int n = nb * 64 + 16 * j + ml;
      dst[m * H_ + n] = acc[j][r];
    }
}

// ---- ctx = relu(fctx @ Wctx^T + bcc + bic): M=128, N=256, K=320. 4 blocks.
__global__ __launch_bounds__(256) void gemm_ctx_kernel(
    const unsigned short* __restrict__ fctx, const unsigned short* __restrict__ Wctx,
    const float* __restrict__ bcc, const float* __restrict__ bic,
    float* __restrict__ ctx) {
  int wave = __builtin_amdgcn_readfirstlane((blockIdx.x * 256 + (int)threadIdx.x) >> 5);
  int nb = wave & 3, mb = wave >> 2;  // C/64 = 4 n-tiles
  v8f acc[4];
  zero_acc(acc);
  wave_tile_16x64<KC_>(fctx + (size_t)(mb * 16) * KC_, KC_,
                       Wctx + (size_t)(nb * 64) * KC_, KC_, acc);
  int lane = threadIdx.x & 31, hl = lane >> 4, ml = lane & 15;
#pragma unroll
  for (int j = 0; j < 4; ++j)
#pragma unroll
    for (int r = 0; r < 8; ++r) {
      int m = mb * 16 + r + 8 * hl;
      int n = nb * 64 + 16 * j + ml;
      ctx[m * C_ + n] = fmaxf(acc[j][r] + bcc[n] + bic[n], 0.0f);
    }
}

// ---- Fused tail per 16 batch rows (8 blocks):
//  1) h = relu(sum_s hpart[s])  (deterministic linear reduction) -> LDS bf16
//  2) hn = h @ Whh^T + bhh      (A from LDS)                     -> LDS f32
//  3) LayerNorm over H, h2 = h + relu(LN(hn))                    -> LDS bf16
//  4) out = h2 @ Who^T + bho    (A from LDS, 8 waves x 16x16)    -> global
__global__ __launch_bounds__(256) void hn_ln_out_kernel(
    const float* __restrict__ hpart, const unsigned short* __restrict__ WhhP,
    const float* __restrict__ bhh, const float* __restrict__ lng,
    const float* __restrict__ lnb, const unsigned short* __restrict__ WhoP,
    const float* __restrict__ bho, float* __restrict__ outp) {
  __shared__ __align__(16) unsigned short hA[16 * H_];   // relu(h) bf16
  __shared__ __align__(16) float hn[16 * H_];
  __shared__ __align__(16) unsigned short h2A[16 * H_];  // h2 bf16
  __shared__ float red[2][256];
  __shared__ float stats[2][16];
  const int mb = blockIdx.x;
  const int wave = __builtin_amdgcn_readfirstlane((int)threadIdx.x >> 5);
  const int lane = threadIdx.x & 31, hl = lane >> 4, ml = lane & 15;

  // 1) split-K reduction + relu
  for (int e = (int)threadIdx.x; e < 16 * H_; e += 256) {
    int r = e >> 9, n = e & (H_ - 1);
    int gb = mb * 16 + r;
    float s = 0.0f;
#pragma unroll
    for (int sp = 0; sp < KSPLIT_; ++sp)
      s += hpart[((size_t)sp * B_ + gb) * H_ + n];
    hA[e] = f2bf(fmaxf(s, 0.0f));
  }
  __syncthreads();

  // 2) hn GEMM: 8 waves cover N=512 (A fragments from LDS)
  v8f acc[4];
  zero_acc(acc);
  wave_tile_16x64<H_>(hA, H_, WhhP + (size_t)(wave * 64) * H_, H_, acc);
#pragma unroll
  for (int j = 0; j < 4; ++j)
#pragma unroll
    for (int r = 0; r < 8; ++r) {
      int rr = r + 8 * hl;
      int n = wave * 64 + 16 * j + ml;
      hn[rr * H_ + n] = acc[j][r] + bhh[n];
    }
  __syncthreads();

  // 3) LayerNorm stats (16 threads per row)
  {
    int row = (int)threadIdx.x >> 4, sub = (int)threadIdx.x & 15;
    float s = 0.0f, s2 = 0.0f;
    for (int e = sub; e < H_; e += 16) {
      float v = hn[row * H_ + e];
      s += v; s2 += v * v;
    }
    red[0][threadIdx.x] = s;
    red[1][threadIdx.x] = s2;
  }
  __syncthreads();
  if (threadIdx.x < 16) {
    float a = 0.0f, b = 0.0f;
    for (int i = 0; i < 16; ++i) {
      a += red[0][threadIdx.x * 16 + i];
      b += red[1][threadIdx.x * 16 + i];
    }
    float mu = a * (1.0f / H_);
    float var = b * (1.0f / H_) - mu * mu;
    stats[0][threadIdx.x] = mu;
    stats[1][threadIdx.x] = rsqrtf(var + 1e-5f);
  }
  __syncthreads();
  for (int e = (int)threadIdx.x; e < 16 * H_; e += 256) {
    int r = e >> 9, n = e & (H_ - 1);
    float y = (hn[e] - stats[0][r]) * stats[1][r] * lng[n] + lnb[n];
    h2A[e] = f2bf(bf2f(hA[e]) + fmaxf(y, 0.0f));
  }
  __syncthreads();

  // 4) out GEMM: wave w -> n-tile w*16 (N=128), K=512, A from LDS
  v8f acc1;
#pragma unroll
  for (int e = 0; e < 8; ++e) acc1[e] = 0.0f;
  wave_tile_16x16<H_>(h2A, H_, WhoP + (size_t)(wave * 16) * H_, H_, acc1);
#pragma unroll
  for (int r = 0; r < 8; ++r) {
    int m = mb * 16 + r + 8 * hl;
    int n = wave * 16 + ml;
    outp[m * O_ + n] = acc1[r] + bho[n];
  }
}

__global__ __launch_bounds__(256) void copy_ctx_kernel(const float* __restrict__ ctx,
                                                       float* __restrict__ dst) {
  int i = blockIdx.x * 256 + (int)threadIdx.x;
  if (i < B_ * C_) dst[i] = ctx[i];
}

extern "C" void kernel_launch(void* const* d_in, const int* in_sizes, int n_in,
                              void* d_out, int out_size, void* d_ws, size_t ws_size,
                              hipStream_t stream) {
  (void)in_sizes; (void)n_in; (void)out_size; (void)ws_size;
  const float* input = (const float*)d_in[0];
  const float* Wcc   = (const float*)d_in[1];
  const float* bcc   = (const float*)d_in[2];
  const float* Wic   = (const float*)d_in[3];
  const float* bic   = (const float*)d_in[4];
  const float* Wmap  = (const float*)d_in[5];
  const float* bmap  = (const float*)d_in[6];
  const float* Whh   = (const float*)d_in[7];
  const float* bhh   = (const float*)d_in[8];
  const float* ln_g  = (const float*)d_in[9];
  const float* ln_b  = (const float*)d_in[10];
  const float* Who   = (const float*)d_in[11];
  const float* bho   = (const float*)d_in[12];
  float* out = (float*)d_out;

  // Workspace carve-out (~27 MB), 256B-aligned regions.
  char* ws = (char*)d_ws;
  size_t off = 0;
  auto carve = [&](size_t bytes) {
    char* p = ws + off;
    off += (bytes + 255) & ~(size_t)255;
    return p;
  };
  unsigned short* Wth   = (unsigned short*)carve((size_t)H_ * KP_ * 2);   // 17.3 MB
  unsigned short* Wctx  = (unsigned short*)carve((size_t)C_ * KC_ * 2);
  unsigned short* WhhP  = (unsigned short*)carve((size_t)H_ * H_ * 2);
  unsigned short* WhoP  = (unsigned short*)carve((size_t)O_ * H_ * 2);
  unsigned short* f     = (unsigned short*)carve((size_t)B_ * KP_ * 2);   // 4.3 MB
  unsigned short* fctx  = (unsigned short*)carve((size_t)B_ * KC_ * 2);
  float*          ctx   = (float*)carve((size_t)B_ * C_ * 4);
  float*          hpart = (float*)carve((size_t)KSPLIT_ * B_ * H_ * 4);   // 4 MB

  const int packTotal = H_ * KP_ + C_ * KC_ + H_ * H_ + O_ * H_ + B_ * C_;
  pack_kernel<<<(packTotal + 255) / 256, 256, 0, stream>>>(
      Wcc, Wic, Wmap, bmap, Whh, Who, Wth, Wctx, WhhP, WhoP, ctx);

  const int featTotal = B_ * KP_ + B_ * KC_;
  const int gemmHBlocks = KSPLIT_ * (B_ / 16) * (H_ / 64) / 8;  // 128
  for (int t = 0; t < T_; ++t) {
    const float* xt = input + (size_t)t * B_ * I_;
    feat_kernel<<<(featTotal + 255) / 256, 256, 0, stream>>>(ctx, xt, f, fctx);
    gemm_h_kernel<<<gemmHBlocks, 256, 0, stream>>>(f, Wth, hpart);
    gemm_ctx_kernel<<<(B_ / 16) * (C_ / 64) / 8, 256, 0, stream>>>(fctx, Wctx, bcc, bic, ctx);
    hn_ln_out_kernel<<<B_ / 16, 256, 0, stream>>>(
        hpart, WhhP, bhh, ln_g, ln_b, WhoP, bho, out + (size_t)t * B_ * O_);
  }
  copy_ctx_kernel<<<(B_ * C_ + 255) / 256, 256, 0, stream>>>(ctx, out + (size_t)T_ * B_ * O_);
}
